// ODEDriftIntegrator_89739046683450
// MI455X (gfx1250) — compile-verified, matrix operand order checked by reading, not run on hardware
//
#include <hip/hip_runtime.h>
#include <hip/hip_bf16.h>

typedef _Float16 v16h __attribute__((ext_vector_type(16)));
typedef float    v8f  __attribute__((ext_vector_type(8)));

#define HID 64
#define HSTEP 0.0025f            /* DT / N_STEPS */
#define KICK 0.01f               /* sigma * sqrt(dt) = 0.1*0.1 */

__device__ __forceinline__ float fast_tanh(float x) {
#if __has_builtin(__builtin_amdgcn_tanhf)
    return __builtin_amdgcn_tanhf(x);        // single V_TANH_F32 (gfx1250 TRANS op)
#else
    float t = __builtin_amdgcn_exp2f(x * 2.8853900817779268f);
    return 1.0f - 2.0f * __builtin_amdgcn_rcpf(t + 1.0f);
#endif
}

// per-lane K index pattern for 16-bit A(16x32)/B(32x16) operands:
// element e (0..15) of the v16h, half hi -> local K
__device__ __forceinline__ int klocal(int e, int hi) {
    return ((e < 8) ? e : e + 8) + 8 * hi;
}

__global__ __launch_bounds__(256, 4) void ode_dopri_wmma_kernel(
    const float* __restrict__ sp,    // [B*S*3] sigma_points
    const float* __restrict__ Wpts,  // [B*S]   W_points
    const float* __restrict__ W1,    // [2*64]
    const float* __restrict__ b1,    // [64]
    const float* __restrict__ W2,    // [64*64]
    const float* __restrict__ b2,    // [64]
    const float* __restrict__ W3,    // [64]
    const float* __restrict__ b3,    // [1]
    float* __restrict__ out,         // [B]
    int nTiles)
{
    __shared__ __align__(16) float ldsB2[HID];
    __shared__ __align__(16) float ldsW3[HID];
    if (threadIdx.x < HID) {
        ldsB2[threadIdx.x] = b2[threadIdx.x];
        ldsW3[threadIdx.x] = W3[threadIdx.x];
    }
    __syncthreads();

    const int lane = threadIdx.x & 31;
    const int m    = lane & 15;      // point-in-tile (both halves own same point)
    const int hi   = lane >> 4;      // wave half -> feature-subset selector
    const int waveId  = blockIdx.x * (blockDim.x >> 5) + (threadIdx.x >> 5);
    const int nWaves  = gridDim.x * (blockDim.x >> 5);

    // ---- A-operand: W2^T, 4 row-tiles (n) x 2 K-chunks, resident in regs ----
    v16h Aop[4][2];
#pragma unroll
    for (int t = 0; t < 4; ++t) {
#pragma unroll
        for (int c = 0; c < 2; ++c) {
            v16h a;
            const int n = t * 16 + m;            // A row M -> hidden-output feature n
#pragma unroll
            for (int e = 0; e < 16; ++e) {
                const int k = 32 * c + klocal(e, hi);
                a[e] = (_Float16)W2[k * HID + n]; // W2^T[n,k] = W2[k,n]
            }
            Aop[t][c] = a;
        }
    }

    // layer-1 slope params (tile-invariant), f32 in regs
    float w1x[2][16];
#pragma unroll
    for (int c = 0; c < 2; ++c)
#pragma unroll
        for (int e = 0; e < 16; ++e)
            w1x[c][e] = W1[32 * c + klocal(e, hi)];

    const float b3v = b3[0];
    const float* pB2 = &ldsB2[8 * hi];   // per-half layer-3 feature base
    const float* pW3 = &ldsW3[8 * hi];

#pragma unroll 1
    for (int tile = waveId; tile < nTiles; tile += nWaves) {
        const int p = tile * 16 + m;
        const float x0 = sp[3 * p + 0];
        const float u  = sp[3 * p + 1];
        const float w  = sp[3 * p + 2];

        // layer-1 offsets: cu[f] = u*W1[1,f] + b1[f]   (per tile)
        float cu[2][16];
#pragma unroll
        for (int c = 0; c < 2; ++c)
#pragma unroll
            for (int e = 0; e < 16; ++e) {
                const int f = 32 * c + klocal(e, hi);
                cu[c][e] = fmaf(u, W1[HID + f], b1[f]);
            }

        float x = x0;

        // one full MLP drift eval: dx = W3 . tanh(W2 . tanh(W1.[x,u]+b1) + b2) + b3
        auto rhs = [&](float xc) -> float {
            asm volatile("" ::: "memory");   // keep per-eval LDS loads in the loop
            v16h h1[2];
#pragma unroll
            for (int c = 0; c < 2; ++c)
#pragma unroll
                for (int e = 0; e < 16; ++e)
                    h1[c][e] = (_Float16)fast_tanh(fmaf(xc, w1x[c][e], cu[c][e]));

            float psum[4];
#pragma unroll
            for (int t = 0; t < 4; ++t) {
                // b2 folded into the WMMA C operand (matches C/D lane layout)
                const float4 b2a = *(const float4*)(pB2 + 16 * t);
                const float4 b2b = *(const float4*)(pB2 + 16 * t + 4);
                v8f acc;
                acc[0] = b2a.x; acc[1] = b2a.y; acc[2] = b2a.z; acc[3] = b2a.w;
                acc[4] = b2b.x; acc[5] = b2b.y; acc[6] = b2b.z; acc[7] = b2b.w;
                acc = __builtin_amdgcn_wmma_f32_16x16x32_f16(
                    false, Aop[t][0], false, h1[0], (short)0, acc, false, false);
                acc = __builtin_amdgcn_wmma_f32_16x16x32_f16(
                    false, Aop[t][1], false, h1[1], (short)0, acc, false, false);

                const float4 w3a = *(const float4*)(pW3 + 16 * t);
                const float4 w3b = *(const float4*)(pW3 + 16 * t + 4);
                float pt;
                pt = fast_tanh(acc[0]) * w3a.x;
                pt = fmaf(fast_tanh(acc[1]), w3a.y, pt);
                pt = fmaf(fast_tanh(acc[2]), w3a.z, pt);
                pt = fmaf(fast_tanh(acc[3]), w3a.w, pt);
                pt = fmaf(fast_tanh(acc[4]), w3b.x, pt);
                pt = fmaf(fast_tanh(acc[5]), w3b.y, pt);
                pt = fmaf(fast_tanh(acc[6]), w3b.z, pt);
                pt = fmaf(fast_tanh(acc[7]), w3b.w, pt);
                psum[t] = pt;
            }
            const float partial = (psum[0] + psum[1]) + (psum[2] + psum[3]);
            // other half holds the complementary 32 hidden features of this point
            const float other = __shfl_xor(partial, 16, 32);
            return partial + other + b3v;
        };

        // 4 fixed Dormand-Prince steps; b2=0 and b7=0 => k7 never needed
#pragma unroll 1
        for (int s = 0; s < 4; ++s) {
            const float k1 = rhs(x);
            const float k2 = rhs(fmaf(HSTEP * 0.2f, k1, x));
            const float k3 = rhs(x + HSTEP * (0.075f * k1 + 0.225f * k2));
            const float k4 = rhs(x + HSTEP * ((44.0f/45.0f)*k1 - (56.0f/15.0f)*k2
                                              + (32.0f/9.0f)*k3));
            const float k5 = rhs(x + HSTEP * ((19372.0f/6561.0f)*k1 - (25360.0f/2187.0f)*k2
                                              + (64448.0f/6561.0f)*k3 - (212.0f/729.0f)*k4));
            const float k6 = rhs(x + HSTEP * ((9017.0f/3168.0f)*k1 - (355.0f/33.0f)*k2
                                              + (46732.0f/5247.0f)*k3 + (49.0f/176.0f)*k4
                                              - (5103.0f/18656.0f)*k5));
            x = x + HSTEP * ((35.0f/384.0f)*k1 + (500.0f/1113.0f)*k3 + (125.0f/192.0f)*k4
                             - (2187.0f/6784.0f)*k5 + (11.0f/84.0f)*k6);
        }

        const float xe = fmaf(KICK, w, x);    // diffusion kick
        if (hi == 0) {                        // one contribution per point
            atomicAdd(&out[p / 5], Wpts[p] * xe);
        }
    }
}

extern "C" void kernel_launch(void* const* d_in, const int* in_sizes, int n_in,
                              void* d_out, int out_size, void* d_ws, size_t ws_size,
                              hipStream_t stream) {
    const float* sp   = (const float*)d_in[0];
    const float* Wpts = (const float*)d_in[1];
    const float* W1   = (const float*)d_in[2];
    const float* b1   = (const float*)d_in[3];
    const float* W2   = (const float*)d_in[4];
    const float* b2   = (const float*)d_in[5];
    const float* W3   = (const float*)d_in[6];
    const float* b3   = (const float*)d_in[7];
    float* out = (float*)d_out;

    const int npts   = in_sizes[1];   // B*S = 655360
    const int nTiles = npts / 16;     // 40960, exact

    hipMemsetAsync(d_out, 0, (size_t)out_size * sizeof(float), stream);

    dim3 grid(1024), block(256);      // 8192 waves, 5 tiles each
    ode_dopri_wmma_kernel<<<grid, block, 0, stream>>>(
        sp, Wpts, W1, b1, W2, b2, W3, b3, out, nTiles);
}
